// MultiAttentionHead_32263794328440
// MI455X (gfx1250) — matrix operand order repaired and checked by reading, not compiled, not run on hardware
//
#include <hip/hip_runtime.h>
#include <stdint.h>

#define B_ 4
#define S_ 1024
#define E_ 1024
#define H_ 16
#define D_ 64
#define NEGV (-1e9f)

typedef __attribute__((ext_vector_type(16))) __bf16        v16bf;
typedef __attribute__((ext_vector_type(8)))  float         v8f;
typedef __attribute__((ext_vector_type(4)))  unsigned int  v4u;

typedef unsigned short ush;

union Frag {
    v16bf v;
    v4u   q[2];
    unsigned int u[8];
    ush   s[16];
};

// fp32 -> bf16 (round to nearest even), pure bit ops
__device__ __forceinline__ ush f2bs(float f) {
    unsigned int u = __float_as_uint(f);
    u += 0x7fffu + ((u >> 16) & 1u);
    return (ush)(u >> 16);
}

// ---------------------------------------------------------------------------
// Kernel 1: x fp32 -> bf16 (row-major), W fp32 -> bf16 TRANSPOSED ([N][K]).
// Transposed W makes every WMMA B-fragment two contiguous 16-byte runs.
// ---------------------------------------------------------------------------
__global__ __launch_bounds__(256) void cvt_kernel(
    const float* __restrict__ x,
    const float* __restrict__ wq, const float* __restrict__ wk, const float* __restrict__ wv,
    ush* __restrict__ xb,
    ush* __restrict__ wtq, ush* __restrict__ wtk, ush* __restrict__ wtv)
{
    int i = blockIdx.x * 256 + threadIdx.x;           // covers B*S*E + 3*E*E exactly
    if (i < B_ * S_ * E_) {
        xb[i] = f2bs(x[i]);
        return;
    }
    int j = i - B_ * S_ * E_;
    int m = j >> 20;                                  // which W (E*E == 1<<20)
    int e = j & ((1 << 20) - 1);
    int k = e >> 10, n = e & (E_ - 1);                // source W[k][n], coalesced reads
    const float* src = (m == 0) ? wq : (m == 1) ? wk : wv;
    ush* dst         = (m == 0) ? wtq : (m == 1) ? wtk : wtv;
    dst[(size_t)n * E_ + k] = f2bs(src[e]);           // W^T[n][k]
}

// ---------------------------------------------------------------------------
// Kernel 2: QKV projection GEMM, no LDS, no barriers.
// Block = 8 waves; wave w owns a 16x16 tile of a 16(M) x 128(N) strip.
// All fragments are 2x contiguous b128 loads (L2-resident operands).
// Q,K stored head-split [B,H,S,D]; V stored TRANSPOSED [B,H,D,S] so the
// attention P@V B-fragment is contiguous as well.
// ---------------------------------------------------------------------------
__global__ __launch_bounds__(256) void proj_kernel(
    const ush* __restrict__ xb,
    const ush* __restrict__ wtq, const ush* __restrict__ wtk, const ush* __restrict__ wtv,
    const float* __restrict__ bq, const float* __restrict__ bk, const float* __restrict__ bv,
    ush* __restrict__ qb, ush* __restrict__ kbuf, ush* __restrict__ vtb)
{
    const int z = blockIdx.z;
    const ush* wt     = (z == 0) ? wtq : (z == 1) ? wtk : wtv;
    const float* bias = (z == 0) ? bq  : (z == 1) ? bk  : bv;

    const int lane = threadIdx.x & 31;
    const int w    = threadIdx.x >> 5;
    const int m0   = blockIdx.x * 16;                 // row tile over B*S rows
    const int n0   = blockIdx.y * 128 + w * 16;       // this wave's N sub-tile

    const int mrow  = lane & 15;
    const int kbase = (lane < 16) ? 0 : 8;            // A-frag K half (ISA 7.12.2)
    const int nn    = lane & 15;
    const int kb2   = (lane < 16) ? 0 : 16;           // B-frag K half
    const int colg  = n0 + nn;

    const ush* arow = xb + (size_t)(m0 + mrow) * E_;  // A: x row (row-major)
    const ush* brow = wt + (size_t)colg * E_;         // B: W^T row (contiguous K)

    v8f c = {};
    for (int k0 = 0; k0 < E_; k0 += 32) {
        Frag a, bm;
        a.q[0]  = *(const v4u*)(arow + k0 + kbase);          // K kbase..kbase+7
        a.q[1]  = *(const v4u*)(arow + k0 + 16 + kbase);     // K 16+kbase..
        bm.q[0] = *(const v4u*)(brow + k0 + kbase);          // K kbase..kbase+7
        bm.q[1] = *(const v4u*)(brow + k0 + 16 + kbase);     // K 16+kbase..
        c = __builtin_amdgcn_wmma_f32_16x16x32_bf16(false, a.v, false, bm.v,
                                                    (short)0, c, false, false);
    }

    // Epilogue: bias, convert, scatter into head-split layouts.
    const float bval = bias[colg];
    const int hi = (lane >> 4) & 1;
    const int h = colg >> 6, d = colg & 63;
#pragma unroll
    for (int r = 0; r < 8; ++r) {
        int rowg = m0 + r + hi * 8;
        int bb = rowg >> 10, ss = rowg & (S_ - 1);
        ush val = f2bs(c[r] + bval);
        if (z == 2) {
            vtb[(((size_t)(bb * H_ + h)) * D_ + d) * S_ + ss] = val;   // V^T [B,H,D,S]
        } else {
            ush* dst = (z == 0) ? qb : kbuf;
            dst[(((size_t)(bb * H_ + h)) * S_ + ss) * D_ + d] = val;   // [B,H,S,D]
        }
    }
}

// ---------------------------------------------------------------------------
// Kernel 3: flash attention, barrier-free. Block = one (b,h) x 128 queries
// (8 independent waves x 16 queries). LDS used only for the wave-private
// P (probabilities) transpose into WMMA A-fragment layout.
// ---------------------------------------------------------------------------
__global__ __launch_bounds__(256) void attn_kernel(
    const ush* __restrict__ qb,
    const ush* __restrict__ kbuf,
    const ush* __restrict__ vtb,
    const int* __restrict__ event_lengths,
    float* __restrict__ out)
{
    __shared__ ush pt[8][16 * 32];                    // per-wave P tile

    const int tid = threadIdx.x, lane = tid & 31, w = tid >> 5;
    const int bh = blockIdx.y;                        // b*H + h
    const int b = bh >> 4, h = bh & 15;
    const int q0 = blockIdx.x * 128 + w * 16;
    const int len = event_lengths[b];

    const int nn   = lane & 15;
    const int hi   = (lane >> 4) & 1;
    const int kb16 = hi * 16;
    const int mrow  = lane & 15;
    const int kbase = (lane < 16) ? 0 : 8;

    const ush* krow = kbuf + (size_t)bh * S_ * D_;    // K [S,D]
    const ush* vrow = vtb  + (size_t)bh * D_ * S_;    // V^T [D,S]

    // Q fragments (both K=32 steps over D=64), 4x b128, kept in registers
    Frag qa[2];
    {
        const ush* qr = qb + ((size_t)bh * S_ + q0 + mrow) * D_;
#pragma unroll
        for (int ds = 0; ds < 2; ++ds) {
            qa[ds].q[0] = *(const v4u*)(qr + ds * 32 + kbase);
            qa[ds].q[1] = *(const v4u*)(qr + ds * 32 + 16 + kbase);
        }
    }

    v8f acc[4] = {};
    float mI[8], lI[8];
#pragma unroll
    for (int r = 0; r < 8; ++r) { mI[r] = -3.0e38f; lI[r] = 0.0f; }

    ush* pw = pt[w];

    for (int t0 = 0; t0 < S_; t0 += 32) {
        if (t0 + 32 < S_) {   // next K/V tiles -> global_prefetch_b8
            __builtin_prefetch(krow + (size_t)(t0 + 32) * D_ + lane * 64, 0, 1);
            __builtin_prefetch(vrow + (size_t)lane * S_ + t0 + 32, 0, 1);
        }

        // Scores: 16 queries x 32 keys as two 16x16 WMMA accumulators.
        // B-fragment = 2x contiguous b128 from K row t (K is [S,D] row-major).
        v8f s0, s1;
#pragma unroll
        for (int j = 0; j < 2; ++j) {
            const ush* krt = krow + (size_t)(t0 + j * 16 + nn) * D_;
            v8f sc = {};
#pragma unroll
            for (int ds = 0; ds < 2; ++ds) {
                Frag bk_;
                bk_.q[0] = *(const v4u*)(krt + ds * 32 + kbase);
                bk_.q[1] = *(const v4u*)(krt + ds * 32 + 16 + kbase);
                sc = __builtin_amdgcn_wmma_f32_16x16x32_bf16(false, qa[ds].v, false, bk_.v,
                                                             (short)0, sc, false, false);
            }
            if (j == 0) s0 = sc; else s1 = sc;
        }

        // Scale, mask, online softmax (each row lives in one half-wave).
        float p0[8], p1[8];
#pragma unroll
        for (int r = 0; r < 8; ++r) {
            int qrow = q0 + r + hi * 8;
            float v0 = s0[r] * 0.125f;                // 1/sqrt(64)
            float v1 = s1[r] * 0.125f;
            bool rowok = qrow < len;
            if (!(rowok && (t0 + nn) < len))      v0 = NEGV;
            if (!(rowok && (t0 + 16 + nn) < len)) v1 = NEGV;
            float rmax = fmaxf(v0, v1);
#pragma unroll
            for (int msk = 1; msk < 16; msk <<= 1)
                rmax = fmaxf(rmax, __shfl_xor(rmax, msk, 16));
            float newm  = fmaxf(mI[r], rmax);
            float alpha = __expf(mI[r] - newm);
            p0[r] = __expf(v0 - newm);
            p1[r] = __expf(v1 - newm);
            float rsum = p0[r] + p1[r];
#pragma unroll
            for (int msk = 1; msk < 16; msk <<= 1)
                rsum += __shfl_xor(rsum, msk, 16);
            lI[r] = lI[r] * alpha + rsum;
            mI[r] = newm;
#pragma unroll
            for (int dt = 0; dt < 4; ++dt) acc[dt][r] *= alpha;
        }

        // P -> wave-private LDS, re-read in WMMA A-fragment layout.
#pragma unroll
        for (int r = 0; r < 8; ++r) {
            int m = r + hi * 8;
            pw[m * 32 + nn]      = f2bs(p0[r]);
            pw[m * 32 + 16 + nn] = f2bs(p1[r]);
        }
        asm volatile("s_wait_dscnt 0" ::: "memory");   // wave-internal LDS RAW

        Frag pa;
        pa.q[0] = *(const v4u*)(pw + mrow * 32 + kbase);
        pa.q[1] = *(const v4u*)(pw + mrow * 32 + 16 + kbase);

        // acc += P(16x32) @ V(32x64): V^T rows give contiguous B-fragments.
#pragma unroll
        for (int dt = 0; dt < 4; ++dt) {
            const ush* vrt = vrow + (size_t)(dt * 16 + nn) * S_ + t0 + kbase;
            Frag vb_;
            vb_.q[0] = *(const v4u*)(vrt);
            vb_.q[1] = *(const v4u*)(vrt + 16);
            acc[dt] = __builtin_amdgcn_wmma_f32_16x16x32_bf16(false, pa.v, false, vb_.v,
                                                              (short)0, acc[dt], false, false);
        }
    }

    // Normalize and store fp32 output [B,S,E] (heads concatenated).
#pragma unroll
    for (int dt = 0; dt < 4; ++dt) {
#pragma unroll
        for (int r = 0; r < 8; ++r) {
            int qrow = q0 + r + hi * 8;
            int col = h * 64 + dt * 16 + nn;
            out[((size_t)(b * S_ + qrow)) * E_ + col] = acc[dt][r] / lI[r];
        }
    }
}

// ---------------------------------------------------------------------------
extern "C" void kernel_launch(void* const* d_in, const int* in_sizes, int n_in,
                              void* d_out, int out_size, void* d_ws, size_t ws_size,
                              hipStream_t stream) {
    (void)in_sizes; (void)n_in; (void)out_size; (void)ws_size;
    const float* x  = (const float*)d_in[0];
    const float* Wq = (const float*)d_in[1];
    const float* bq = (const float*)d_in[2];
    const float* Wk = (const float*)d_in[3];
    const float* bk = (const float*)d_in[4];
    const float* Wv = (const float*)d_in[5];
    const float* bv = (const float*)d_in[6];
    const int*   el = (const int*)d_in[7];
    float* out = (float*)d_out;

    // Workspace (bf16 as ushort), ~38 MB total, fully L2-resident:
    ush* xb  = (ush*)d_ws;                            // [B*S, E]        8 MB
    ush* wtq = xb  + (size_t)B_ * S_ * E_;            // W_q^T [E, E]    2 MB
    ush* wtk = wtq + (size_t)E_ * E_;                 //                 2 MB
    ush* wtv = wtk + (size_t)E_ * E_;                 //                 2 MB
    ush* qb  = wtv + (size_t)E_ * E_;                 // Q [B,H,S,D]     8 MB
    ush* kb  = qb  + (size_t)B_ * S_ * E_;            // K [B,H,S,D]     8 MB
    ush* vt  = kb  + (size_t)B_ * S_ * E_;            // V^T [B,H,D,S]   8 MB

    const int cvt_elems = B_ * S_ * E_ + 3 * E_ * E_; // 7340032 = 28672*256
    cvt_kernel<<<dim3(cvt_elems / 256), dim3(256), 0, stream>>>(
        x, Wq, Wk, Wv, xb, wtq, wtk, wtv);

    proj_kernel<<<dim3((B_ * S_) / 16, E_ / 128, 3), dim3(256), 0, stream>>>(
        xb, wtq, wtk, wtv, bq, bk, bv, qb, kb, vt);

    attn_kernel<<<dim3(S_ / 128, B_ * H_), dim3(256), 0, stream>>>(qb, kb, vt, el, out);
}
